// GeomMoEBlock_43327630082143
// MI455X (gfx1250) — matrix operand order repaired and checked by reading, not compiled, not run on hardware
//
#include <hip/hip_runtime.h>
#include <hip/hip_bf16.h>

// ---------------- types ----------------
typedef _Float16 v16h __attribute__((ext_vector_type(16)));
typedef float    v8f  __attribute__((ext_vector_type(8)));
typedef unsigned int u32x4 __attribute__((ext_vector_type(4)));

union Frag16 { u32x4 q[2]; v16h v; };

// ---------------- problem constants ----------------
#define N_NODES 65536
#define DIM     256
#define NEXP    8
#define HDIM    1024
#define TM      64      // node tile per workgroup
#define CHUNK   64      // Hd chunk
#define HSTR    264     // H_lds row stride (halves) -> 528B, conflict-free b128 reads
#define CSTR    72      // h_lds row stride (halves) -> 144B, conflict-free b128 reads
#define XSTR    260     // x_lds row stride (floats) -> 1040B

#define W1T_ELEMS (NEXP * (DIM/32) * (HDIM/16) * 512)   // 2,097,152 halves
#define W2T_ELEMS (NEXP * (HDIM/32) * (DIM/16) * 512)   // 2,097,152 halves

// ============================================================
// Kernel 1: gating softmax. One wave per node (wave32).
// ============================================================
__global__ __launch_bounds__(256) void gate_kernel(
    const float* __restrict__ H, const float* __restrict__ gw,
    const float* __restrict__ gb, float* __restrict__ wts)
{
    const int lane = threadIdx.x & 31;
    const int wave = threadIdx.x >> 5;
    const int node = blockIdx.x * 8 + wave;

    float p[8];
#pragma unroll
    for (int e = 0; e < 8; ++e) p[e] = 0.f;

#pragma unroll
    for (int i = 0; i < 8; ++i) {
        const int d = lane + i * 32;
        const float h = H[node * DIM + d];
#pragma unroll
        for (int e = 0; e < 8; ++e) p[e] += h * gw[d * 8 + e];
    }
#pragma unroll
    for (int off = 16; off >= 1; off >>= 1) {
#pragma unroll
        for (int e = 0; e < 8; ++e) p[e] += __shfl_xor(p[e], off, 32);
    }
    float m = -1e30f;
#pragma unroll
    for (int e = 0; e < 8; ++e) { p[e] += gb[e]; m = fmaxf(m, p[e]); }
    float s = 0.f;
#pragma unroll
    for (int e = 0; e < 8; ++e) { p[e] = __expf(p[e] - m); s += p[e]; }
    const float inv = 1.f / s;
    if (lane == 0) {
#pragma unroll
        for (int e = 0; e < 8; ++e) wts[node * 8 + e] = p[e] * inv;
    }
}

// ============================================================
// Kernel 2/3: pre-swizzle weights (f32 row-major -> f16 WMMA
// B-fragment tiles). Tile = 32x16 (KxN), 512 halves = 1KB:
//   element [lane*16 + j] = W[k = kb*32 + (lane&16 ? 16:0) + j,
//                             n = ct*16 + (lane&15)]
// so the GEMM loads each lane's 32B contiguously.
// ============================================================
__global__ __launch_bounds__(256) void w1_transform(
    const float* __restrict__ W1, _Float16* __restrict__ w1t)
{
    const int g = blockIdx.x * 256 + threadIdx.x;     // 0 .. W1T_ELEMS-1
    const int j    =  g        & 15;
    const int lane = (g >> 4)  & 31;
    const int ct   = (g >> 9)  & 63;                  // Hd/16 tiles
    const int kb   = (g >> 15) & 7;                   // D/32 k-blocks
    const int e    =  g >> 18;
    const int k = kb * 32 + ((lane & 16) ? 16 : 0) + j;
    const int n = ct * 16 + (lane & 15);
    w1t[g] = (_Float16)W1[(e * DIM + k) * HDIM + n];
}

__global__ __launch_bounds__(256) void w2_transform(
    const float* __restrict__ W2, _Float16* __restrict__ w2t)
{
    const int g = blockIdx.x * 256 + threadIdx.x;     // 0 .. W2T_ELEMS-1
    const int j    =  g        & 15;
    const int lane = (g >> 4)  & 31;
    const int ct   = (g >> 9)  & 15;                  // D/16 tiles
    const int kb   = (g >> 13) & 31;                  // Hd/32 k-blocks
    const int e    =  g >> 18;
    const int k = kb * 32 + ((lane & 16) ? 16 : 0) + j;   // h index
    const int n = ct * 16 + (lane & 15);                  // d index
    w2t[g] = (_Float16)W2[(e * HDIM + k) * DIM + n];
}

// ============================================================
// Kernel 4: fused MoE + residual + LayerNorm.
// Grid: N/64 blocks, 256 threads (8 waves).
// Wave w: row-block rb = w&3 (16 rows), col-half ch = w>>2.
// ============================================================
__global__ __launch_bounds__(256) void moe_kernel(
    const float* __restrict__ H,     const float* __restrict__ bias1,
    const float* __restrict__ bias2, const float* __restrict__ lng,
    const float* __restrict__ lnb,   const float* __restrict__ wts,
    const _Float16* __restrict__ w1t, const _Float16* __restrict__ w2t,
    float* __restrict__ out)
{
    extern __shared__ char smem[];
    _Float16* Hl = (_Float16*)smem;                       // [64][HSTR] f16
    _Float16* hl = (_Float16*)(smem + TM * HSTR * 2);     // [64][CSTR] f16
    float*    xl = (float*)smem;                          // [64][XSTR] f32 (aliases Hl/hl)

    const int tid  = threadIdx.x;
    const int lane = tid & 31;
    const int wave = tid >> 5;
    const int rb   = wave & 3;        // row block (16 rows)
    const int ch   = wave >> 2;       // column half (128 cols)
    const int node0 = blockIdx.x * TM;

    const int arow = lane & 15;               // A-frag row within block
    const int asel = (lane >> 4) << 4;        // 0 or 16: K sub-offset, contiguous k = 16h + s
    const int crow = (lane >> 4) << 3;        // C-frag row base offset (0 or 8)

    // ---- stage H tile into LDS as f16 (coalesced) ----
    for (int i = tid; i < TM * DIM; i += 256) {
        const int r = i >> 8, c = i & 255;
        Hl[r * HSTR + c] = (_Float16)H[(node0 + r) * DIM + c];
    }
    __syncthreads();

    // ---- f32 output accumulator: 8 C tiles (16x16) per wave ----
    v8f acc[8];
#pragma unroll
    for (int jj = 0; jj < 8; ++jj)
#pragma unroll
        for (int v = 0; v < 8; ++v) acc[jj][v] = 0.f;

    for (int e = 0; e < NEXP; ++e) {
        // gate weights for this wave's 16 rows (indexed by C-frag vgpr)
        float wrow[8];
#pragma unroll
        for (int v = 0; v < 8; ++v)
            wrow[v] = wts[(node0 + rb * 16 + crow + v) * 8 + e];

        for (int c = 0; c < HDIM / CHUNK; ++c) {
            // prefetch next chunk's first W1 tile (global_prefetch_b8)
            if (c + 1 < HDIM / CHUNK)
                __builtin_prefetch(w1t + (((e * 8) * 64) + ((c + 1) * 4 + ch)) * 512 + lane * 16, 0, 0);

            // ---------- GEMM1: h = relu(H @ W1[:, chunk] + b1) * w ----------
            // wave computes col tiles j = ch and j = ch+2 of the 64-wide chunk
            v8f hc0, hc1;
#pragma unroll
            for (int v = 0; v < 8; ++v) { hc0[v] = 0.f; hc1[v] = 0.f; }

#pragma unroll
            for (int kb = 0; kb < 8; ++kb) {
                Frag16 a;
                const int ko = kb * 32 + asel;
                const _Float16* ap = Hl + (rb * 16 + arow) * HSTR + ko;
                a.q[0] = *(const u32x4*)(const void*)ap;
                a.q[1] = *(const u32x4*)(const void*)(ap + 8);

                const _Float16* bp0 = w1t + (((e * 8 + kb) * 64) + (c * 4 + ch)) * 512 + lane * 16;
                Frag16 b0, b1f;
                b0.q[0]  = ((const u32x4*)(const void*)bp0)[0];
                b0.q[1]  = ((const u32x4*)(const void*)bp0)[1];
                const _Float16* bp1 = bp0 + 2 * 512;     // col tile j = ch+2
                b1f.q[0] = ((const u32x4*)(const void*)bp1)[0];
                b1f.q[1] = ((const u32x4*)(const void*)bp1)[1];

                hc0 = __builtin_amdgcn_wmma_f32_16x16x32_f16(false, a.v, false, b0.v,
                                                             (short)0, hc0, false, false);
                hc1 = __builtin_amdgcn_wmma_f32_16x16x32_f16(false, a.v, false, b1f.v,
                                                             (short)0, hc1, false, false);
            }

            // epilogue: +b1, relu, scale by gate weight, stage f16 to LDS
            {
                const int hcol0 = c * 64 + ch * 16 + (lane & 15);
                const float bA = bias1[e * HDIM + hcol0];
                const float bB = bias1[e * HDIM + hcol0 + 32];
                const int lr = rb * 16 + crow;
#pragma unroll
                for (int v = 0; v < 8; ++v) {
                    float x0 = hc0[v] + bA; x0 = x0 > 0.f ? x0 : 0.f; x0 *= wrow[v];
                    float x1 = hc1[v] + bB; x1 = x1 > 0.f ? x1 : 0.f; x1 *= wrow[v];
                    hl[(lr + v) * CSTR + (ch * 16 + (lane & 15))]       = (_Float16)x0;
                    hl[(lr + v) * CSTR + ((ch + 2) * 16 + (lane & 15))] = (_Float16)x1;
                }
            }
            __syncthreads();

            // ---------- GEMM2: acc += h_chunk @ W2[chunk, :] ----------
#pragma unroll
            for (int k2 = 0; k2 < 2; ++k2) {
                Frag16 a;
                const int ko = k2 * 32 + asel;
                const _Float16* ap = hl + (rb * 16 + arow) * CSTR + ko;
                a.q[0] = *(const u32x4*)(const void*)ap;
                a.q[1] = *(const u32x4*)(const void*)(ap + 8);
#pragma unroll
                for (int jj = 0; jj < 8; ++jj) {
                    const _Float16* bp =
                        w2t + ((e * 32 + (c * 2 + k2)) * 16 + (ch * 8 + jj)) * 512 + lane * 16;
                    Frag16 b;
                    b.q[0] = ((const u32x4*)(const void*)bp)[0];
                    b.q[1] = ((const u32x4*)(const void*)bp)[1];
                    acc[jj] = __builtin_amdgcn_wmma_f32_16x16x32_f16(false, a.v, false, b.v,
                                                                     (short)0, acc[jj], false, false);
                }
            }
            __syncthreads();
        }
    }

    // ---- epilogue: + sum_e w[n,e]*b2[e,:], + residual H; stage x to LDS ----
    const int col0 = ch * 128 + (lane & 15);
#pragma unroll
    for (int v = 0; v < 8; ++v) {
        const int lrow = rb * 16 + crow + v;
        const int node = node0 + lrow;
        float we[8];
#pragma unroll
        for (int e = 0; e < 8; ++e) we[e] = wts[node * 8 + e];
#pragma unroll
        for (int jj = 0; jj < 8; ++jj) {
            const int col = col0 + jj * 16;
            float s = 0.f;
#pragma unroll
            for (int e = 0; e < 8; ++e) s += we[e] * bias2[e * DIM + col];
            xl[lrow * XSTR + col] = acc[jj][v] + s + H[node * DIM + col];
        }
    }
    __syncthreads();

    // ---- LayerNorm: each wave owns 8 rows; shfl reduction over 32 lanes ----
    for (int rr = 0; rr < 8; ++rr) {
        const int row = wave * 8 + rr;
        float s = 0.f, s2 = 0.f, xv[8];
#pragma unroll
        for (int i = 0; i < 8; ++i) {
            const float x = xl[row * XSTR + lane + i * 32];
            xv[i] = x; s += x; s2 += x * x;
        }
#pragma unroll
        for (int off = 16; off >= 1; off >>= 1) {
            s  += __shfl_xor(s,  off, 32);
            s2 += __shfl_xor(s2, off, 32);
        }
        const float mu  = s * (1.f / 256.f);
        const float var = s2 * (1.f / 256.f) - mu * mu;
        const float rs  = rsqrtf(var + 1e-5f);
#pragma unroll
        for (int i = 0; i < 8; ++i) {
            const int col = lane + i * 32;
            out[(node0 + row) * DIM + col] = (xv[i] - mu) * rs * lng[col] + lnb[col];
        }
    }
}

// ============================================================
// Launcher
// ============================================================
extern "C" void kernel_launch(void* const* d_in, const int* in_sizes, int n_in,
                              void* d_out, int out_size, void* d_ws, size_t ws_size,
                              hipStream_t stream) {
    const float* H   = (const float*)d_in[0];
    const float* gw  = (const float*)d_in[1];
    const float* gb  = (const float*)d_in[2];
    const float* W1  = (const float*)d_in[3];
    const float* b1  = (const float*)d_in[4];
    const float* W2  = (const float*)d_in[5];
    const float* b2  = (const float*)d_in[6];
    const float* lng = (const float*)d_in[7];
    const float* lnb = (const float*)d_in[8];
    float* out = (float*)d_out;

    char* ws = (char*)d_ws;
    float*    wts = (float*)ws;                                   // N*8 f32  (2 MB)
    _Float16* w1t = (_Float16*)(ws + (size_t)N_NODES * 8 * 4);    // 4 MB
    _Float16* w2t = w1t + W1T_ELEMS;                              // 4 MB

    gate_kernel<<<N_NODES / 8, 256, 0, stream>>>(H, gw, gb, wts);
    w1_transform<<<W1T_ELEMS / 256, 256, 0, stream>>>(W1, w1t);
    w2_transform<<<W2T_ELEMS / 256, 256, 0, stream>>>(W2, w2t);

    const size_t ldsBytes = (size_t)TM * XSTR * 4;                // 66,560 B (> GEMM-phase usage)
    moe_kernel<<<N_NODES / TM, 256, ldsBytes, stream>>>(
        H, b1, b2, lng, lnb, wts, w1t, w2t, out);
}
